// StateTransformerBlock_13632226197805
// MI455X (gfx1250) — matrix operand order repaired and checked by reading, not compiled, not run on hardware
//
#include <hip/hip_runtime.h>
#include <hip/hip_bf16.h>

// ---------------------------------------------------------------------------
// Types / helpers for CDNA5 WMMA (wave32, 16x16x32 bf16 -> f32)
// ---------------------------------------------------------------------------
typedef __attribute__((ext_vector_type(16))) __bf16 v16bf;
typedef __attribute__((ext_vector_type(8)))  __bf16 v8bf;
typedef __attribute__((ext_vector_type(8)))  float  v8f;

__device__ __forceinline__ __bf16 f2bf(float f) {
  unsigned u = __builtin_bit_cast(unsigned, f);
  u += 0x7FFFu + ((u >> 16) & 1u);               // round-to-nearest-even
  unsigned short s = (unsigned short)(u >> 16);
  return __builtin_bit_cast(__bf16, s);
}
__device__ __forceinline__ float bf2f(__bf16 h) {
  unsigned short s = __builtin_bit_cast(unsigned short, h);
  unsigned u = ((unsigned)s) << 16;
  return __builtin_bit_cast(float, u);
}
__device__ __forceinline__ v16bf pack16(v8bf lo, v8bf hi) {
  v16bf r;
#pragma unroll
  for (int i = 0; i < 8; ++i) { r[i] = lo[i]; r[i + 8] = hi[i]; }
  return r;
}
__device__ __forceinline__ v8f zero8() {
  v8f z = {0.f, 0.f, 0.f, 0.f, 0.f, 0.f, 0.f, 0.f};
  return z;
}
__device__ __forceinline__ v8f wmma_bf16(v16bf a, v16bf b, v8f c) {
  return __builtin_amdgcn_wmma_f32_16x16x32_bf16(false, a, false, b, (short)0, c,
                                                 false, false);
}

// ---------------------------------------------------------------------------
// gfx1250 async global->LDS copy (ASYNCcnt-tracked), with sync fallback.
// Builtin signature (from toolchain diagnostic): arg0 is a pointer to a
// 16-byte int vector in the global (AS1) address space; arg1 the LDS (AS3)
// destination; args 2/3 are imm offset / cpol.
// ---------------------------------------------------------------------------
#if defined(__has_builtin)
#if __has_builtin(__builtin_amdgcn_global_load_async_to_lds_b128)
#define HAVE_ASYNC_LDS 1
#endif
#endif
#ifndef HAVE_ASYNC_LDS
#define HAVE_ASYNC_LDS 0
#endif

#if HAVE_ASYNC_LDS
typedef int v4i_vs __attribute__((vector_size(4 * sizeof(int))));
__device__ __forceinline__ void async_ld16(const void* g, void* l) {
  void* gnc = (void*)g;  // drop const
  __builtin_amdgcn_global_load_async_to_lds_b128(
      (__attribute__((address_space(1))) v4i_vs*)gnc,
      (__attribute__((address_space(3))) v4i_vs*)l, 0, 0);
}
__device__ __forceinline__ void wait_async0() {
#if __has_builtin(__builtin_amdgcn_s_wait_asynccnt)
  __builtin_amdgcn_s_wait_asynccnt(0);
#else
  asm volatile("s_wait_asynccnt 0x0" ::: "memory");
#endif
}
#endif

// Problem constants
constexpr int CB  = 4;     // batch
constexpr int CSM = 1024;  // state length
constexpr int CSI = 4096;  // input length
constexpr int CD  = 256;   // model dim
constexpr int CH  = 8;     // heads
constexpr int CFF = 1024;  // ffn dim
constexpr int M1  = CB * CSM;   // 4096
constexpr int M2  = CB * CSI;   // 16384

// ---------------------------------------------------------------------------
// Generic WMMA GEMM:  out = act(A[M,K] * W[K,N] + bias[N]) (* mul[M,N])
// Wave computes 16x64; block = 8 waves = 128 rows; grid.y tiles N/64; grid.z
// batches. The shared 32x64 W tile is staged in LDS (double-buffered, async
// copies when available) so all 8 waves reuse one copy; A fragments are two
// contiguous 8-element runs per lane straight from global.
// ---------------------------------------------------------------------------
template <bool ABF16, bool RELU, bool MUL, bool OUTF, bool OUTB>
__global__ __launch_bounds__(256) void gemm_wmma_kernel(
    const void* __restrict__ Av, long strideAz,
    const float* __restrict__ W, long strideWz,
    const float* __restrict__ bias, long strideBz,
    const float* __restrict__ mul,
    float* __restrict__ outF, __bf16* __restrict__ outB, long strideOz,
    int M, int N, int K) {
  __shared__ alignas(16) float wt[2][32][68];   // 32 K-rows x 64 cols (+4 pad)

  const int tid  = threadIdx.x;
  const int lane = tid & 31;
  const int wv   = tid >> 5;
  const int half = lane >> 4;
  const int l16  = lane & 15;
  const int row0 = blockIdx.x * 128 + wv * 16;
  const int col0 = blockIdx.y * 64;
  const int z    = blockIdx.z;
  (void)M;

  const float* Wz = W + (size_t)z * strideWz;
  const float* bz = bias + (size_t)z * strideBz;

  auto stageW = [&](int kb, int s) {
#pragma unroll
    for (int r = 0; r < 2; ++r) {
      const int idx = r * 256 + tid;      // 512 chunks of 4 floats = 32x64
      const int rw  = idx >> 4;
      const int cc  = (idx & 15) << 2;
      const float* src = Wz + (size_t)(kb * 32 + rw) * N + (col0 + cc);
#if HAVE_ASYNC_LDS
      async_ld16(src, &wt[s][rw][cc]);
#else
      *(float4*)&wt[s][rw][cc] = *(const float4*)src;
#endif
    }
  };

  v8f acc[4];
#pragma unroll
  for (int c = 0; c < 4; ++c) acc[c] = zero8();

  const int KB = K >> 5;
  const int rowA = row0 + l16;

  stageW(0, 0);
  for (int kb = 0; kb < KB; ++kb) {
    const int s = kb & 1;
#if HAVE_ASYNC_LDS
    wait_async0();
#endif
    __syncthreads();
    if (kb + 1 < KB) stageW(kb + 1, s ^ 1);

    v16bf afrag;
    const int k0 = kb * 32 + half * 8;
    if (ABF16) {
      const __bf16* A = (const __bf16*)Av + (size_t)z * strideAz +
                        (size_t)rowA * K + k0;
      afrag = pack16(*(const v8bf*)A, *(const v8bf*)(A + 16));
    } else {
      const float* A = (const float*)Av + (size_t)z * strideAz +
                       (size_t)rowA * K + k0;
#pragma unroll
      for (int e = 0; e < 8; ++e) {
        afrag[e]     = f2bf(A[e]);
        afrag[e + 8] = f2bf(A[16 + e]);
      }
    }
#pragma unroll
    for (int c = 0; c < 4; ++c) {
      v16bf bfrag;
#pragma unroll
      for (int e = 0; e < 16; ++e)
        bfrag[e] = f2bf(wt[s][half * 8 + e + (e & 8)][c * 16 + l16]);
      acc[c] = wmma_bf16(afrag, bfrag, acc[c]);
    }
  }

#pragma unroll
  for (int c = 0; c < 4; ++c) {
    const int col = col0 + c * 16 + l16;
    const float bb = bz[col];
#pragma unroll
    for (int j = 0; j < 8; ++j) {
      const int row = row0 + j + half * 8;
      float v = acc[c][j] + bb;
      if (RELU) v = fmaxf(v, 0.0f);
      if (MUL) v *= mul[(size_t)row * N + col];
      const size_t oidx = (size_t)z * strideOz + (size_t)row * N + col;
      if (OUTF) outF[oidx] = v;
      if (OUTB) outB[oidx] = f2bf(v);
    }
  }
}

// ---------------------------------------------------------------------------
// Flash-attention kernel (multi-query: shared K/V across heads).
// grid = (SM/64, B, H), block = 128 (4 waves, each owns 16 query rows).
// K tiles are double-buffered in LDS with the next tile's copy issued (async
// when available) while the current one feeds QK^T WMMAs. V is transposed on
// store so both QK^T and PV fragments read contiguous LDS. The per-row self
// score against mk/mv is folded in after the stream.
// ---------------------------------------------------------------------------
__global__ __launch_bounds__(128) void attn_wmma_kernel(
    const __bf16* __restrict__ qbuf,   // [H][B*SM][D]
    const __bf16* __restrict__ ikbuf,  // [B*SI][D]
    const __bf16* __restrict__ ivbuf,  // [B*SI][D]
    const __bf16* __restrict__ mkbuf,  // [B*SM][D]
    const __bf16* __restrict__ mvbuf,  // [B*SM][D]
    __bf16* __restrict__ concat)       // [B*SM][H*D]
{
  __shared__ alignas(16) __bf16 kt[2][32][264]; // 32 keys x 256 (+8), x2 stages
  __shared__ alignas(16) __bf16 vt[256][40];    // transposed: 256 d x 32 keys
  __shared__ alignas(16) __bf16 pt[4][16][40];  // per-wave P staging

  const int tid  = threadIdx.x;
  const int lane = tid & 31;
  const int wv   = tid >> 5;
  const int half = lane >> 4;
  const int l16  = lane & 15;
  const int h    = blockIdx.z;
  const int b    = blockIdx.y;
  const int q0   = blockIdx.x * 64 + wv * 16;
  constexpr int NB = CSI / 32;

  const __bf16* ikb = ikbuf + (size_t)b * CSI * CD;
  const __bf16* ivb = ivbuf + (size_t)b * CSI * CD;

  auto stageK = [&](int jb, int s) {
#pragma unroll
    for (int r = 0; r < 8; ++r) {
      const int idx = r * 128 + tid;     // 1024 chunks of 8 bf16 = 32x256
      const int key = idx >> 5;
      const int d0  = (idx & 31) * 8;
      const __bf16* src = ikb + (size_t)(jb * 32 + key) * CD + d0;
#if HAVE_ASYNC_LDS
      async_ld16(src, &kt[s][key][d0]);
#else
      *(v8bf*)&kt[s][key][d0] = *(const v8bf*)src;
#endif
    }
  };
  auto loadV = [&](int jb) {
#pragma unroll
    for (int r = 0; r < 8; ++r) {
      const int idx = r * 128 + tid;
      const int key = idx >> 5;
      const int d0  = (idx & 31) * 8;
      const size_t src = (size_t)(jb * 32 + key) * CD + d0;
      v8bf vv = *(const v8bf*)(ivb + src);
#pragma unroll
      for (int i = 0; i < 8; ++i) vt[d0 + i][key] = vv[i];
      if (jb + 1 < NB) __builtin_prefetch(ivb + src + (size_t)32 * CD, 0, 1);
    }
  };

  // Preload all Q A-fragments for this wave's 16 rows (D=256 -> 8 k-blocks).
  v16bf aq[8];
  {
    const __bf16* qp = qbuf + ((size_t)h * M1 + (size_t)b * CSM + q0 + l16) * CD +
                       half * 16;
#pragma unroll
    for (int kb = 0; kb < 8; ++kb)
      aq[kb] = pack16(*(const v8bf*)(qp + kb * 32),
                      *(const v8bf*)(qp + kb * 32 + 8));
  }

  v8f o[16];
#pragma unroll
  for (int c = 0; c < 16; ++c) o[c] = zero8();
  float mrow[8], lrow[8];
#pragma unroll
  for (int j = 0; j < 8; ++j) { mrow[j] = -1e30f; lrow[j] = 0.0f; }

  stageK(0, 0);
  for (int jb = 0; jb < NB; ++jb) {
    const int s = jb & 1;
#if HAVE_ASYNC_LDS
    wait_async0();
#endif
    __syncthreads();                    // kt[s] ready; vt reads of jb-1 done
    if (jb + 1 < NB) stageK(jb + 1, s ^ 1);
    loadV(jb);
    __syncthreads();                    // vt ready

    // S = Q * K^T : two 16x16 tiles (keys 0-15, 16-31), K-dim = D = 8 blocks.
    v8f s0 = zero8(), s1 = zero8();
#pragma unroll
    for (int kb = 0; kb < 8; ++kb) {
      const __bf16* kp0 = &kt[s][l16][kb * 32 + half * 16];
      s0 = wmma_bf16(aq[kb], pack16(*(const v8bf*)kp0, *(const v8bf*)(kp0 + 8)),
                     s0);
      const __bf16* kp1 = &kt[s][16 + l16][kb * 32 + half * 16];
      s1 = wmma_bf16(aq[kb], pack16(*(const v8bf*)kp1, *(const v8bf*)(kp1 + 8)),
                     s1);
    }

    // Online softmax. Row r = j + 8*half lives across the lane's 16-lane half,
    // so xor-masks 1..8 reduce each row independently.
#pragma unroll
    for (int j = 0; j < 8; ++j) {
      const float x0 = s0[j] * 0.0625f;  // 1/sqrt(256)
      const float x1 = s1[j] * 0.0625f;
      float bm = fmaxf(x0, x1);
#pragma unroll
      for (int off = 8; off >= 1; off >>= 1)
        bm = fmaxf(bm, __shfl_xor(bm, off, 32));
      const float mnew = fmaxf(mrow[j], bm);
      const float corr = __expf(mrow[j] - mnew);
      const float p0 = __expf(x0 - mnew);
      const float p1 = __expf(x1 - mnew);
      float ps = p0 + p1;
#pragma unroll
      for (int off = 8; off >= 1; off >>= 1) ps += __shfl_xor(ps, off, 32);
      lrow[j] = lrow[j] * corr + ps;
      mrow[j] = mnew;
#pragma unroll
      for (int c = 0; c < 16; ++c) o[c][j] *= corr;
      pt[wv][j + half * 8][l16]      = f2bf(p0);
      pt[wv][j + half * 8][16 + l16] = f2bf(p1);
    }

    // O += P(16x32) * V(32x256): P through LDS into A-layout; V from vt.
    const __bf16* pp = &pt[wv][l16][half * 16];
    const v16bf pa = pack16(*(const v8bf*)pp, *(const v8bf*)(pp + 8));
#pragma unroll
    for (int c = 0; c < 16; ++c) {
      const __bf16* vp = &vt[c * 16 + l16][half * 16];
      o[c] = wmma_bf16(pa, pack16(*(const v8bf*)vp, *(const v8bf*)(vp + 8)),
                       o[c]);
    }
  }

  // Fold in the per-row self key (mk/mv), normalize, store concat (bf16).
  const __bf16* mkb = mkbuf + (size_t)b * CSM * CD;
  const __bf16* mvb = mvbuf + (size_t)b * CSM * CD;
  const __bf16* qb2 = qbuf + ((size_t)h * M1 + (size_t)b * CSM) * CD;
#pragma unroll
  for (int j = 0; j < 8; ++j) {
    const int m = q0 + j + half * 8;
    const __bf16* qr = qb2 + (size_t)m * CD + l16 * 16;
    const __bf16* kr = mkb + (size_t)m * CD + l16 * 16;
    float part = 0.0f;
#pragma unroll
    for (int t = 0; t < 16; ++t) part += bf2f(qr[t]) * bf2f(kr[t]);
#pragma unroll
    for (int off = 8; off >= 1; off >>= 1) part += __shfl_xor(part, off, 32);
    const float ss    = part * 0.0625f;
    const float mnew  = fmaxf(mrow[j], ss);
    const float corr  = __expf(mrow[j] - mnew);
    const float pself = __expf(ss - mnew);
    const float linv  = 1.0f / (lrow[j] * corr + pself);
    __bf16* op = concat + ((size_t)b * CSM + m) * (CH * CD) + (size_t)h * CD;
    const __bf16* mvr = mvb + (size_t)m * CD;
#pragma unroll
    for (int c = 0; c < 16; ++c) {
      const int dcol = c * 16 + l16;
      const float v = (o[c][j] * corr + pself * bf2f(mvr[dcol])) * linv;
      op[dcol] = f2bf(v);
    }
  }
}

// ---------------------------------------------------------------------------
// Residual + LayerNorm over D=256; one wave per row (8 rows / 256-thread block)
// ---------------------------------------------------------------------------
__global__ __launch_bounds__(256) void ln_kernel(
    const float* __restrict__ a, const float* __restrict__ res,
    const float* __restrict__ g, const float* __restrict__ bta,
    float* __restrict__ outF, __bf16* __restrict__ outB) {
  const int lane = threadIdx.x & 31;
  const int wv   = threadIdx.x >> 5;
  const size_t row = (size_t)blockIdx.x * 8 + wv;
  const float* pa = a + row * CD;
  const float* pr = res + row * CD;
  float x[8];
  float s = 0.f;
#pragma unroll
  for (int i = 0; i < 8; ++i) {
    const int col = i * 32 + lane;
    x[i] = pa[col] + pr[col];
    s += x[i];
  }
#pragma unroll
  for (int off = 16; off >= 1; off >>= 1) s += __shfl_xor(s, off, 32);
  const float mean = s * (1.0f / 256.0f);
  float vs = 0.f;
#pragma unroll
  for (int i = 0; i < 8; ++i) { const float d = x[i] - mean; vs += d * d; }
#pragma unroll
  for (int off = 16; off >= 1; off >>= 1) vs += __shfl_xor(vs, off, 32);
  const float rstd = rsqrtf(vs * (1.0f / 256.0f) + 1e-6f);
#pragma unroll
  for (int i = 0; i < 8; ++i) {
    const int col = i * 32 + lane;
    const float y = (x[i] - mean) * rstd * g[col] + bta[col];
    outF[row * CD + col] = y;
    if (outB) outB[row * CD + col] = f2bf(y);
  }
}

// ---------------------------------------------------------------------------
// Workspace layout (bytes). Total ~98 MiB.
// ---------------------------------------------------------------------------
constexpr size_t SZ_IK = (size_t)M2 * CD * 2;         // 8 MiB (bf16)
constexpr size_t SZ_MK = (size_t)M1 * CD * 2;         // 2 MiB
constexpr size_t SZ_Q  = (size_t)CH * M1 * CD * 2;    // 16 MiB
constexpr size_t SZ_CC = (size_t)M1 * CH * CD * 2;    // 16 MiB
constexpr size_t SZ_F  = (size_t)M1 * CD * 4;         // 4 MiB (f32 D-wide)
constexpr size_t SZ_HB = (size_t)M1 * CFF * 2;        // 8 MiB (bf16 FF-wide)
constexpr size_t SZ_VL = (size_t)M1 * CFF * 4;        // 16 MiB (f32 FF-wide)

constexpr size_t OFF_IK = 0;
constexpr size_t OFF_IV = OFF_IK + SZ_IK;
constexpr size_t OFF_MK = OFF_IV + SZ_IK;
constexpr size_t OFF_MV = OFF_MK + SZ_MK;
constexpr size_t OFF_Q  = OFF_MV + SZ_MK;
constexpr size_t OFF_CC = OFF_Q + SZ_Q;
constexpr size_t OFF_AO = OFF_CC + SZ_CC;   // attn_out f32
constexpr size_t OFF_X  = OFF_AO + SZ_F;    // x f32
constexpr size_t OFF_XB = OFF_X + SZ_F;     // x bf16
constexpr size_t OFF_H  = OFF_XB + SZ_MK;   // h bf16
constexpr size_t OFF_VL = OFF_H + SZ_HB;    // h@Wg_l f32
constexpr size_t OFF_G  = OFF_VL + SZ_VL;   // gated bf16
constexpr size_t OFF_FF = OFF_G + SZ_HB;    // ff f32

extern "C" void kernel_launch(void* const* d_in, const int* in_sizes, int n_in,
                              void* d_out, int out_size, void* d_ws,
                              size_t ws_size, hipStream_t stream) {
  (void)in_sizes; (void)n_in; (void)out_size; (void)ws_size;
  const float* state = (const float*)d_in[0];
  const float* input = (const float*)d_in[1];
  const float* Wk    = (const float*)d_in[2];
  const float* bk    = (const float*)d_in[3];
  const float* Wv    = (const float*)d_in[4];
  const float* bv    = (const float*)d_in[5];
  const float* Wq    = (const float*)d_in[6];
  const float* bq    = (const float*)d_in[7];
  const float* Wo    = (const float*)d_in[8];
  const float* bo    = (const float*)d_in[9];
  const float* ln1g  = (const float*)d_in[10];
  const float* ln1b  = (const float*)d_in[11];
  const float* Win   = (const float*)d_in[12];
  const float* bin   = (const float*)d_in[13];
  const float* Wgnl  = (const float*)d_in[14];
  const float* bgnl  = (const float*)d_in[15];
  const float* Wgl   = (const float*)d_in[16];
  const float* bgl   = (const float*)d_in[17];
  const float* Wgo   = (const float*)d_in[18];
  const float* bgo   = (const float*)d_in[19];
  const float* ln2g  = (const float*)d_in[20];
  const float* ln2b  = (const float*)d_in[21];
  float* out = (float*)d_out;

  char* ws = (char*)d_ws;
  __bf16* ik  = (__bf16*)(ws + OFF_IK);
  __bf16* iv  = (__bf16*)(ws + OFF_IV);
  __bf16* mk  = (__bf16*)(ws + OFF_MK);
  __bf16* mv  = (__bf16*)(ws + OFF_MV);
  __bf16* qb  = (__bf16*)(ws + OFF_Q);
  __bf16* cc  = (__bf16*)(ws + OFF_CC);
  float*  ao  = (float*)(ws + OFF_AO);
  float*  x   = (float*)(ws + OFF_X);
  __bf16* xb  = (__bf16*)(ws + OFF_XB);
  __bf16* hb  = (__bf16*)(ws + OFF_H);
  float*  vl  = (float*)(ws + OFF_VL);
  __bf16* gb  = (__bf16*)(ws + OFF_G);
  float*  ff  = (float*)(ws + OFF_FF);

  const dim3 blk(256);

  // K/V projections of input_seq and state_seq (bf16 outputs for WMMA).
  gemm_wmma_kernel<false, false, false, false, true>
      <<<dim3(M2 / 128, CD / 64, 1), blk, 0, stream>>>(
          input, 0, Wk, 0, bk, 0, nullptr, nullptr, ik, 0, M2, CD, CD);
  gemm_wmma_kernel<false, false, false, false, true>
      <<<dim3(M2 / 128, CD / 64, 1), blk, 0, stream>>>(
          input, 0, Wv, 0, bv, 0, nullptr, nullptr, iv, 0, M2, CD, CD);
  gemm_wmma_kernel<false, false, false, false, true>
      <<<dim3(M1 / 128, CD / 64, 1), blk, 0, stream>>>(
          state, 0, Wk, 0, bk, 0, nullptr, nullptr, mk, 0, M1, CD, CD);
  gemm_wmma_kernel<false, false, false, false, true>
      <<<dim3(M1 / 128, CD / 64, 1), blk, 0, stream>>>(
          state, 0, Wv, 0, bv, 0, nullptr, nullptr, mv, 0, M1, CD, CD);

  // Per-head queries, batched over grid.z = H.
  gemm_wmma_kernel<false, false, false, false, true>
      <<<dim3(M1 / 128, CD / 64, CH), blk, 0, stream>>>(
          state, 0, Wq, (long)CD * CD, bq, CD, nullptr, nullptr, qb,
          (long)M1 * CD, M1, CD, CD);

  // Flash attention (with self key), writes concat bf16 [B*SM, H*D].
  attn_wmma_kernel<<<dim3(CSM / 64, CB, CH), dim3(128), 0, stream>>>(
      qb, ik, iv, mk, mv, cc);

  // attn_out = concat @ Wo + bo  (A bf16, K = H*D = 2048)
  gemm_wmma_kernel<true, false, false, true, false>
      <<<dim3(M1 / 128, CD / 64, 1), blk, 0, stream>>>(
          cc, 0, Wo, 0, bo, 0, nullptr, ao, nullptr, 0, M1, CD, CH * CD);

  // x = LN1(attn_out + state) -> f32 + bf16 copies
  ln_kernel<<<dim3(M1 / 8), blk, 0, stream>>>(ao, state, ln1g, ln1b, x, xb);

  // h = relu(x @ W_in + b_in) -> bf16 [M1, FF]
  gemm_wmma_kernel<true, true, false, false, true>
      <<<dim3(M1 / 128, CFF / 64, 1), blk, 0, stream>>>(
          xb, 0, Win, 0, bin, 0, nullptr, nullptr, hb, 0, M1, CFF, CD);

  // vl = h @ Wg_l + bg_l (linear branch, f32)
  gemm_wmma_kernel<true, false, false, true, false>
      <<<dim3(M1 / 128, CFF / 64, 1), blk, 0, stream>>>(
          hb, 0, Wgl, 0, bgl, 0, nullptr, vl, nullptr, 0, M1, CFF, CFF);

  // g = relu(h @ Wg_nl + bg_nl) * vl -> bf16
  gemm_wmma_kernel<true, true, true, false, true>
      <<<dim3(M1 / 128, CFF / 64, 1), blk, 0, stream>>>(
          hb, 0, Wgnl, 0, bgnl, 0, vl, nullptr, gb, 0, M1, CFF, CFF);

  // ff = g @ Wg_o + bg_o -> f32
  gemm_wmma_kernel<true, false, false, true, false>
      <<<dim3(M1 / 128, CD / 64, 1), blk, 0, stream>>>(
          gb, 0, Wgo, 0, bgo, 0, nullptr, ff, nullptr, 0, M1, CD, CFF);

  // out = LN2(ff + x)
  ln_kernel<<<dim3(M1 / 8), blk, 0, stream>>>(ff, x, ln2g, ln2b, out, nullptr);
}
